// ConfidenceLossV2_70300024701559
// MI455X (gfx1250) — compile-verified, matrix operand order checked by reading, not run on hardware
//
#include <hip/hip_runtime.h>
#include <hip/hip_bf16.h>

typedef __attribute__((ext_vector_type(16))) _Float16 v16h;
typedef __attribute__((ext_vector_type(8)))  float    v8f;

#define NSEG   64
#define NBATCH 8
#define CFEAT  64
#define HE     128
#define WE     128
#define NPIX   (HE * WE)        /* 16384 feature pixels per batch */
#define HW     (512 * 512)      /* full-res pixels per batch      */
#define BLOCKS_PER_BATCH 32
#define PIX_PER_BLOCK    (NPIX / BLOCKS_PER_BATCH)   /* 512 */

/* workspace layout (floats) */
#define WS_SUM     0            /* [512] sum of per-pixel mean err per bin */
#define WS_CNT     512          /* [512] pixel counts per bin              */
#define WS_POS     1024         /* [512] positive-mask pixel counts        */
#define WS_SCALARS 1536         /* [0]=sum(mse*m), [1]=sum(m)              */
#define WS_TOTAL   1538

__global__ __launch_bounds__(256)
void zero_ws_kernel(float* __restrict__ ws) {
    int i = blockIdx.x * 256 + threadIdx.x;
    if (i < WS_TOTAL) ws[i] = 0.0f;
}

/* Segment statistics at feature resolution.
 * Each wave reduces 16 pixels x 64 channels of (enc-dec)^2 with two
 * v_wmma_f32_16x16x32_f16 ops (B = ones), then bins results into LDS. */
__global__ __launch_bounds__(256)
void seg_stats_kernel(const float* __restrict__ enc,
                      const float* __restrict__ dec,
                      const float* __restrict__ masks,
                      const int*   __restrict__ segs,
                      float*       __restrict__ ws)
{
    __shared__ float sSum[NSEG];
    __shared__ float sCnt[NSEG];
    __shared__ float sPos[NSEG];

    const int tid  = threadIdx.x;
    const int b    = blockIdx.x / BLOCKS_PER_BATCH;
    const int blk  = blockIdx.x % BLOCKS_PER_BATCH;

    if (tid < NSEG) { sSum[tid] = 0.0f; sCnt[tid] = 0.0f; sPos[tid] = 0.0f; }
    __syncthreads();

    const int lane = tid & 31;
    const int wave = tid >> 5;
    const int kHi  = (lane >> 4) * 8;   /* high half-wave handles K+8 sub-blocks */
    const int mrow = lane & 15;         /* A-matrix row M for this lane          */

    const size_t featBase = (size_t)b * CFEAT * NPIX;

    v16h onesB;
    #pragma unroll
    for (int e = 0; e < 16; ++e) onesB[e] = (_Float16)1.0f;

    #pragma unroll 1
    for (int it = 0; it < PIX_PER_BLOCK / 128; ++it) {
        const int    pixBase = blk * PIX_PER_BLOCK + it * 128 + wave * 16;
        const size_t pix     = (size_t)(pixBase + mrow);

        /* hint next tile into cache (global_prefetch_b8) */
        __builtin_prefetch(&enc[featBase + pix + 128], 0, 1);
        __builtin_prefetch(&dec[featBase + pix + 128], 0, 1);

        v8f acc = {};
        #pragma unroll
        for (int cb = 0; cb < CFEAT; cb += 32) {
            /* Build A (16x32 f16): element e of v16h holds K = kHi + (e<8 ? e : e+8) */
            v16h a;
            #pragma unroll
            for (int e = 0; e < 16; ++e) {
                const int    k   = cb + kHi + ((e < 8) ? e : (e + 8));
                const size_t off = featBase + (size_t)k * NPIX + pix;
                const float  d   = enc[off] - dec[off];
                a[e] = (_Float16)(d * d);
            }
            acc = __builtin_amdgcn_wmma_f32_16x16x32_f16(
                      false, a, false, onesB, (short)0, acc, false, false);
        }

        /* D layout: lane L holds D[M, N=L] with M = acc-index (+8 for hi half).
         * Every column is the row-sum; lanes 0-7 emit pixels 0-7, lanes 16-23
         * emit pixels 8-15. */
        if ((lane & 15) < 8) {
            float s;
            switch (lane & 7) {
                case 0: s = acc[0]; break;  case 1: s = acc[1]; break;
                case 2: s = acc[2]; break;  case 3: s = acc[3]; break;
                case 4: s = acc[4]; break;  case 5: s = acc[5]; break;
                case 6: s = acc[6]; break;  default: s = acc[7]; break;
            }
            const int   m       = (lane < 16) ? (lane & 7) : (8 + (lane & 7));
            const int   pp      = pixBase + m;             /* feature pixel   */
            const float meanErr = s * (1.0f / (float)CFEAT);

            const int    yy   = pp >> 7;
            const int    xx   = pp & (WE - 1);
            const size_t full = (size_t)b * HW + (size_t)(yy * 4) * 512 + (size_t)(xx * 4);

            const int   sg = segs[full] & (NSEG - 1);
            const float mk = masks[full];

            atomicAdd(&sSum[sg], meanErr);
            atomicAdd(&sCnt[sg], 1.0f);
            if (mk > 0.0f && mk < 0.5f) atomicAdd(&sPos[sg], 1.0f);
        }
    }
    __syncthreads();

    if (tid < NSEG && sCnt[tid] != 0.0f) {
        const int bin = b * NSEG + tid;
        atomicAdd(&ws[WS_SUM + bin], sSum[tid]);
        atomicAdd(&ws[WS_CNT + bin], sCnt[tid]);
        atomicAdd(&ws[WS_POS + bin], sPos[tid]);
    }
}

/* Recovery loss over [8,3,512,512]: 4 pixels per thread via b128 loads
 * (512 B fetched per VMEM instruction per wave), wave32 shuffle reduction,
 * 2 global atomics per block. */
__global__ __launch_bounds__(256)
void recov_loss_kernel(const float* __restrict__ outputs,
                       const float* __restrict__ inputs,
                       const float* __restrict__ masks,
                       float*       __restrict__ ws)
{
    __shared__ float w0[8];
    __shared__ float w1[8];

    const int    tid  = threadIdx.x;
    const int    lane = tid & 31;
    const int    wave = tid >> 5;
    const size_t g4   = (size_t)blockIdx.x * 256 + tid;  /* quad index          */
    const size_t p4   = g4 * 4;                          /* < NBATCH*HW exactly */
    const int    b    = (int)(p4 / HW);
    const int    p    = (int)(p4 % HW);                  /* multiple of 4       */

    const float4 mk4 = *(const float4*)&masks[(size_t)b * HW + p];
    const float  mk[4] = { mk4.x, mk4.y, mk4.z, mk4.w };

    float mse[4] = { 0.0f, 0.0f, 0.0f, 0.0f };
    const size_t base = (size_t)b * 3 * HW + p;
    #pragma unroll
    for (int c = 0; c < 3; ++c) {
        const float4 in4 = *(const float4*)&inputs[base + (size_t)c * HW];
        const float4 o4  = *(const float4*)&outputs[base + (size_t)c * HW];
        const float  in[4] = { in4.x, in4.y, in4.z, in4.w };
        const float  o [4] = { o4.x,  o4.y,  o4.z,  o4.w  };
        #pragma unroll
        for (int i = 0; i < 4; ++i) {
            const float t = (mk[i] >= 0.5f) ? 0.0f : in[i];
            const float d = o[i] - t;
            mse[i] += d * d;
        }
    }

    float v0 = 0.0f, v1 = 0.0f;
    #pragma unroll
    for (int i = 0; i < 4; ++i) {
        const float m = (mk[i] > 0.0f) ? 1.0f : 0.0f;
        v0 += mse[i] * m;
        v1 += m;
    }

    /* wave32 reduction */
    #pragma unroll
    for (int off = 16; off > 0; off >>= 1) {
        v0 += __shfl_down(v0, off, 32);
        v1 += __shfl_down(v1, off, 32);
    }
    if (lane == 0) { w0[wave] = v0; w1[wave] = v1; }
    __syncthreads();
    if (tid == 0) {
        float a0 = 0.0f, a1 = 0.0f;
        #pragma unroll
        for (int i = 0; i < 8; ++i) { a0 += w0[i]; a1 += w1[i]; }
        atomicAdd(&ws[WS_SCALARS + 0], a0);
        atomicAdd(&ws[WS_SCALARS + 1], a1);
    }
}

__global__ __launch_bounds__(512)
void finalize_kernel(const float* __restrict__ ws, float* __restrict__ out)
{
    __shared__ float r0[512];
    __shared__ float r1[512];

    const int t = threadIdx.x;                 /* one thread per bin (512) */
    const float cnt  = ws[WS_CNT + t];
    const float sum  = ws[WS_SUM + t];
    const float pos  = ws[WS_POS + t];
    const float safe = fmaxf(cnt, 1.0f);
    const float mean = sum / safe;
    const bool  valid = (cnt / (float)NPIX) >= 0.01f;
    const bool  posf  = (pos / safe) > 0.01f;
    const float sel   = (valid && posf) ? 1.0f : 0.0f;

    r0[t] = mean * sel;
    r1[t] = sel;
    __syncthreads();
    #pragma unroll
    for (int s = 256; s > 0; s >>= 1) {
        if (t < s) { r0[t] += r0[t + s]; r1[t] += r1[t + s]; }
        __syncthreads();
    }
    if (t == 0) {
        const float flat_pos_mean = r0[0] / fmaxf(r1[0], 1.0f);
        const float loss_recov    = ws[WS_SCALARS + 0] / fmaxf(ws[WS_SCALARS + 1], 1.0f);
        out[0] = loss_recov + flat_pos_mean;
    }
}

extern "C" void kernel_launch(void* const* d_in, const int* in_sizes, int n_in,
                              void* d_out, int out_size, void* d_ws, size_t ws_size,
                              hipStream_t stream) {
    (void)in_sizes; (void)n_in; (void)out_size; (void)ws_size;

    const float* outputs = (const float*)d_in[0];
    const float* inputs  = (const float*)d_in[1];
    const float* enc1    = (const float*)d_in[2];
    const float* dec1    = (const float*)d_in[3];
    const float* masks   = (const float*)d_in[4];
    const int*   segs    = (const int*)d_in[5];
    /* d_in[6..8]: confidence / iteration / epoch — unused by the loss value */

    float* ws  = (float*)d_ws;
    float* out = (float*)d_out;

    zero_ws_kernel<<<(WS_TOTAL + 255) / 256, 256, 0, stream>>>(ws);
    seg_stats_kernel<<<NBATCH * BLOCKS_PER_BATCH, 256, 0, stream>>>(enc1, dec1, masks, segs, ws);
    recov_loss_kernel<<<(NBATCH * HW) / (256 * 4), 256, 0, stream>>>(outputs, inputs, masks, ws);
    finalize_kernel<<<1, 512, 0, stream>>>(ws, out);
}
